// SpeechT5Attention_37976100831937
// MI455X (gfx1250) — compile-verified, matrix operand order checked by reading, not run on hardware
//
#include <hip/hip_runtime.h>
#include <hip/hip_bf16.h>

typedef __attribute__((ext_vector_type(16))) _Float16 v16h;
typedef __attribute__((ext_vector_type(4)))  _Float16 v4h;
typedef __attribute__((ext_vector_type(8)))  float    v8f;
typedef __attribute__((ext_vector_type(4)))  int      v4i;

#define Bsz 2
#define Ssz 2048
#define Hsz 1024
#define NHsz 16
#define HDsz 64
#define QSCALE 0.125f

#define FST 72   // f16 LDS tile stride (64 + 8 pad) -> 144 B, 16B aligned rows
#define SST 65   // f32 score tile stride (odd -> conflict-free row scans)

// ---------------------------------------------------------------------------
// 16-byte global -> LDS copy. Prefer the CDNA5 async-to-LDS path (ASYNCcnt,
// GLOBAL_LOAD_ASYNC_TO_LDS_B128); fall back to VGPR-bounced b128 copy.
// Builtin signature (from hipcc diagnostic): (int4 AS1*, int4 AS3*, imm, imm)
// ---------------------------------------------------------------------------
#if defined(__has_builtin)
#if __has_builtin(__builtin_amdgcn_global_load_async_to_lds_b128)
#define HAVE_ASYNC_LDS 1
#endif
#endif

typedef __attribute__((address_space(1))) v4i* gv4i_p;
typedef __attribute__((address_space(3))) v4i* lv4i_p;

__device__ __forceinline__ void cp16(void* lds, const void* glob) {
#ifdef HAVE_ASYNC_LDS
  __builtin_amdgcn_global_load_async_to_lds_b128(
      (gv4i_p)(glob), (lv4i_p)(lds), 0, 0);
#else
  *(uint4*)lds = *(const uint4*)glob;
#endif
}

__device__ __forceinline__ void cp_fence() {
#ifdef HAVE_ASYNC_LDS
#if defined(__has_builtin) && __has_builtin(__builtin_amdgcn_s_wait_asynccnt)
  __builtin_amdgcn_s_wait_asynccnt(0);
#else
  asm volatile("s_wait_asynccnt 0" ::: "memory");
#endif
#endif
}

// Stage a 64-row x 64-half tile into LDS (row stride FST halves).
// gstride = global row stride in halves. 128 threads, 4 x 16B each.
__device__ __forceinline__ void stage64(_Float16* __restrict__ lds,
                                        const _Float16* __restrict__ gbase,
                                        long gstride, int tid) {
#pragma unroll
  for (int i = tid; i < 512; i += 128) {
    int row = i >> 3, q = i & 7;
    cp16(lds + row * FST + q * 8, gbase + (long)row * gstride + q * 8);
  }
}

// ---------------------------------------------------------------------------
// Fragment gather from LDS, 16-bit A/B matrix layout (ISA 7.12.2, wave32):
// lane m = lane&15 selects the row (A: M-row, B: N-column of the N x K tile),
// hi = lane>>4 selects K-halves; element e -> K = kofs + 8*hi + (e<8 ? e : e+8).
// Pairs are K-consecutive -> lowers to two ds_load_b128 per fragment.
// ---------------------------------------------------------------------------
__device__ __forceinline__ v16h frag16(const _Float16* __restrict__ base,
                                       int stride, int kofs) {
  const int lane = threadIdx.x & 31;
  const int m  = lane & 15;
  const int hi = lane >> 4;
  const _Float16* r = base + m * stride + kofs + hi * 8;
  v16h f;
#pragma unroll
  for (int e = 0; e < 16; e += 2) {
    int k = (e < 8) ? e : (e + 8);
    f[e]     = r[k];
    f[e + 1] = r[k + 1];
  }
  return f;
}

#define WMMA(a, b, c) __builtin_amdgcn_wmma_f32_16x16x32_f16( \
    false, (a), false, (b), (short)0, (c), false, false)

// ---------------------------------------------------------------------------
// fp32 -> f16 conversion, 4 elements/thread
// ---------------------------------------------------------------------------
__global__ void cvt_f32_f16(const float* __restrict__ src,
                            _Float16* __restrict__ dst, int n) {
  int i = (blockIdx.x * 256 + threadIdx.x) * 4;
  if (i < n) {
    float4 f = *(const float4*)(src + i);
    v4h h;
    h[0] = (_Float16)f.x; h[1] = (_Float16)f.y;
    h[2] = (_Float16)f.z; h[3] = (_Float16)f.w;
    *(v4h*)(dst + i) = h;
  }
}

// ---------------------------------------------------------------------------
// Projection GEMM: out = (A @ W^T + bias) * scale, f16 output.
// vmode=0: per-head [B,NH,S,HD];  vmode=1: transposed per-head [B,NH,HD,S].
// Block = 128 threads (4 waves), tile 64(M) x 64(N), K-step 64.
// ---------------------------------------------------------------------------
__global__ void __launch_bounds__(128) gemm_proj(
    const _Float16* __restrict__ A, const _Float16* __restrict__ W,
    const float* __restrict__ bias, _Float16* __restrict__ dst,
    float scale, int vmode) {
  __shared__ alignas(16) _Float16 As[64 * FST];
  __shared__ alignas(16) _Float16 Bs[64 * FST];
  const int tid = threadIdx.x;
  const int wave = tid >> 5;
  const int lane = tid & 31;
  const int laneN = lane & 15, laneHi = lane >> 4;
  const int m0 = blockIdx.x * 64;
  const int n0 = blockIdx.y * 64;
  const _Float16* Abase = A + (long)m0 * Hsz;
  const _Float16* Wbase = W + (long)n0 * Hsz;

  v8f acc[4] = {};
  for (int k0 = 0; k0 < Hsz; k0 += 64) {
    __syncthreads();
    stage64(As, Abase + k0, Hsz, tid);
    stage64(Bs, Wbase + k0, Hsz, tid);
    cp_fence();
    __syncthreads();
    v16h a0 = frag16(As + wave * 16 * FST, FST, 0);
    v16h a1 = frag16(As + wave * 16 * FST, FST, 32);
#pragma unroll
    for (int nt = 0; nt < 4; ++nt) {
      v16h b0 = frag16(Bs + nt * 16 * FST, FST, 0);
      v16h b1 = frag16(Bs + nt * 16 * FST, FST, 32);
      acc[nt] = WMMA(a0, b0, acc[nt]);
      acc[nt] = WMMA(a1, b1, acc[nt]);
    }
  }
#pragma unroll
  for (int nt = 0; nt < 4; ++nt) {
    int col = n0 + nt * 16 + laneN;
    int h = col >> 6, d = col & 63;
    float bcol = bias[col];
#pragma unroll
    for (int r = 0; r < 8; ++r) {
      int row = m0 + wave * 16 + laneHi * 8 + r;
      int b = row >> 11, s = row & 2047;
      float v = (acc[nt][r] + bcol) * scale;
      long idx = vmode ? ((long)(b * NHsz + h) * HDsz + d) * Ssz + s
                       : ((long)(b * NHsz + h) * Ssz + s) * HDsz + d;
      dst[idx] = (_Float16)v;
    }
  }
}

// ---------------------------------------------------------------------------
// Output projection GEMM: d_out = A @ Wo^T + bo  (fp32 output, [4096,1024])
// ---------------------------------------------------------------------------
__global__ void __launch_bounds__(128) gemm_out(
    const _Float16* __restrict__ A, const _Float16* __restrict__ W,
    const float* __restrict__ bias, float* __restrict__ dst) {
  __shared__ alignas(16) _Float16 As[64 * FST];
  __shared__ alignas(16) _Float16 Bs[64 * FST];
  const int tid = threadIdx.x;
  const int wave = tid >> 5;
  const int lane = tid & 31;
  const int laneN = lane & 15, laneHi = lane >> 4;
  const int m0 = blockIdx.x * 64;
  const int n0 = blockIdx.y * 64;
  const _Float16* Abase = A + (long)m0 * Hsz;
  const _Float16* Wbase = W + (long)n0 * Hsz;

  v8f acc[4] = {};
  for (int k0 = 0; k0 < Hsz; k0 += 64) {
    __syncthreads();
    stage64(As, Abase + k0, Hsz, tid);
    stage64(Bs, Wbase + k0, Hsz, tid);
    cp_fence();
    __syncthreads();
    v16h a0 = frag16(As + wave * 16 * FST, FST, 0);
    v16h a1 = frag16(As + wave * 16 * FST, FST, 32);
#pragma unroll
    for (int nt = 0; nt < 4; ++nt) {
      v16h b0 = frag16(Bs + nt * 16 * FST, FST, 0);
      v16h b1 = frag16(Bs + nt * 16 * FST, FST, 32);
      acc[nt] = WMMA(a0, b0, acc[nt]);
      acc[nt] = WMMA(a1, b1, acc[nt]);
    }
  }
#pragma unroll
  for (int nt = 0; nt < 4; ++nt) {
    int col = n0 + nt * 16 + laneN;
    float bcol = bias[col];
#pragma unroll
    for (int r = 0; r < 8; ++r) {
      int row = m0 + wave * 16 + laneHi * 8 + r;
      dst[(long)row * Hsz + col] = acc[nt][r] + bcol;
    }
  }
}

// ---------------------------------------------------------------------------
// Flash attention: one block per (b,h, 64-row q-tile); 128 threads (4 waves).
// Scores in LDS; online softmax; V arrives pre-transposed ([B,NH,HD,S]).
// ---------------------------------------------------------------------------
__global__ void __launch_bounds__(128) attn_flash(
    const _Float16* __restrict__ q16, const _Float16* __restrict__ k16,
    const _Float16* __restrict__ vt16, const float* __restrict__ pbias,
    const float* __restrict__ mask, _Float16* __restrict__ out16) {
  __shared__ alignas(16) _Float16 Qs[64 * FST];
  __shared__ alignas(16) _Float16 Ks[64 * FST];
  __shared__ alignas(16) _Float16 Vt[64 * FST];
  __shared__ alignas(16) _Float16 Pt[64 * FST];
  __shared__ float Sc[64 * SST];
  __shared__ float rowM[64], rowL[64], rowF[64];

  const int tid = threadIdx.x;
  const int wave = tid >> 5;
  const int lane = tid & 31;
  const int laneN = lane & 15, laneHi = lane >> 4;
  const int qt = blockIdx.x;          // q tile 0..31
  const int bh = blockIdx.y;          // 0..31
  const int b = bh >> 4, h = bh & 15;
  const long headBase = (long)bh * Ssz * HDsz;   // for q16/k16 slices
  const _Float16* Vhead = vt16 + headBase;       // [HD, S] rows for this head

  stage64(Qs, q16 + headBase + (long)qt * 64 * HDsz, HDsz, tid);
  if (tid < 64) { rowM[tid] = -3.0e38f; rowL[tid] = 0.0f; }
  cp_fence();
  __syncthreads();

  const v16h qa0 = frag16(Qs + wave * 16 * FST, FST, 0);
  const v16h qa1 = frag16(Qs + wave * 16 * FST, FST, 32);
  v8f o[4] = {};

  for (int kt = 0; kt < Ssz / 64; ++kt) {
    // stage K tile rows [key][d] and V^T tile rows [d][key]
    stage64(Ks, k16 + headBase + (long)kt * 64 * HDsz, HDsz, tid);
    stage64(Vt, Vhead + kt * 64, Ssz, tid);
    cp_fence();
    __syncthreads();

    // S = Q K^T  (each wave: 16 q-rows x 64 keys)
#pragma unroll
    for (int nt = 0; nt < 4; ++nt) {
      v8f c = {};
      v16h bf0 = frag16(Ks + nt * 16 * FST, FST, 0);
      v16h bf1 = frag16(Ks + nt * 16 * FST, FST, 32);
      c = WMMA(qa0, bf0, c);
      c = WMMA(qa1, bf1, c);
#pragma unroll
      for (int r = 0; r < 8; ++r)
        Sc[(wave * 16 + laneHi * 8 + r) * SST + nt * 16 + laneN] = c[r];
    }
    __syncthreads();

    // add position bias + additive mask (float4 coalesced reads)
    {
      const float4* pb4 =
          (const float4*)(pbias + ((long)h * Ssz + qt * 64) * Ssz + kt * 64);
      const float4* mk4 =
          (const float4*)(mask + ((long)b * Ssz + qt * 64) * Ssz + kt * 64);
#pragma unroll
      for (int i = tid; i < 1024; i += 128) {
        int row = i >> 4, c4 = i & 15;
        float4 pv = pb4[(long)row * (Ssz / 4) + c4];
        float4 mv = mk4[(long)row * (Ssz / 4) + c4];
        float* srow = Sc + row * SST + c4 * 4;
        srow[0] += pv.x + mv.x;
        srow[1] += pv.y + mv.y;
        srow[2] += pv.z + mv.z;
        srow[3] += pv.w + mv.w;
      }
    }
    __syncthreads();

    // online softmax: one thread per q-row
    if (tid < 64) {
      float* srow = Sc + tid * SST;
      float rm = srow[0];
#pragma unroll 8
      for (int j = 1; j < 64; ++j) rm = fmaxf(rm, srow[j]);
      float mOld = rowM[tid];
      float mNew = fmaxf(mOld, rm);
      float f = __expf(mOld - mNew);
      float s = 0.0f;
      _Float16* prow = Pt + tid * FST;
#pragma unroll 8
      for (int j = 0; j < 64; ++j) {
        float p = __expf(srow[j] - mNew);
        s += p;
        prow[j] = (_Float16)p;
      }
      rowL[tid] = rowL[tid] * f + s;
      rowM[tid] = mNew;
      rowF[tid] = f;
    }
    __syncthreads();

    // rescale running O, then O += P @ V
#pragma unroll
    for (int nt = 0; nt < 4; ++nt)
#pragma unroll
      for (int r = 0; r < 8; ++r)
        o[nt][r] *= rowF[wave * 16 + laneHi * 8 + r];

    v16h pa0 = frag16(Pt + wave * 16 * FST, FST, 0);
    v16h pa1 = frag16(Pt + wave * 16 * FST, FST, 32);
#pragma unroll
    for (int nt = 0; nt < 4; ++nt) {
      v16h vb0 = frag16(Vt + nt * 16 * FST, FST, 0);
      v16h vb1 = frag16(Vt + nt * 16 * FST, FST, 32);
      o[nt] = WMMA(pa0, vb0, o[nt]);
      o[nt] = WMMA(pa1, vb1, o[nt]);
    }
    __syncthreads();
  }

  // normalize and write to [B, S, H] f16
#pragma unroll
  for (int nt = 0; nt < 4; ++nt) {
    int d = nt * 16 + laneN;
#pragma unroll
    for (int r = 0; r < 8; ++r) {
      int rowT = wave * 16 + laneHi * 8 + r;
      int s = qt * 64 + rowT;
      float val = o[nt][r] / rowL[rowT];
      out16[((long)(b * Ssz + s)) * Hsz + h * HDsz + d] = (_Float16)val;
    }
  }
}

// ---------------------------------------------------------------------------
extern "C" void kernel_launch(void* const* d_in, const int* in_sizes, int n_in,
                              void* d_out, int out_size, void* d_ws, size_t ws_size,
                              hipStream_t stream) {
  const float* hs  = (const float*)d_in[0];
  const float* pb  = (const float*)d_in[1];
  const float* msk = (const float*)d_in[2];
  const float* Wq  = (const float*)d_in[3];
  const float* bq  = (const float*)d_in[4];
  const float* Wk  = (const float*)d_in[5];
  const float* bk  = (const float*)d_in[6];
  const float* Wv  = (const float*)d_in[7];
  const float* bv  = (const float*)d_in[8];
  const float* Wo  = (const float*)d_in[9];
  const float* bo  = (const float*)d_in[10];
  float* out = (float*)d_out;

  const int NHS = Bsz * Ssz * Hsz;   // 4,194,304
  const int WSZ = Hsz * Hsz;         // 1,048,576
  _Float16* hs16 = (_Float16*)d_ws;
  _Float16* w16  = hs16 + NHS;       // 4 weight matrices
  _Float16* q16  = w16 + 4 * (size_t)WSZ;
  _Float16* k16  = q16 + NHS;
  _Float16* vt16 = k16 + NHS;        // V stored [B,NH,HD,S]
  _Float16* ao16 = vt16 + NHS;

  // fp32 -> f16 staging (4 elems/thread)
  cvt_f32_f16<<<(NHS / 4 + 255) / 256, 256, 0, stream>>>(hs, hs16, NHS);
  cvt_f32_f16<<<(WSZ / 4 + 255) / 256, 256, 0, stream>>>(Wq, w16 + 0 * (size_t)WSZ, WSZ);
  cvt_f32_f16<<<(WSZ / 4 + 255) / 256, 256, 0, stream>>>(Wk, w16 + 1 * (size_t)WSZ, WSZ);
  cvt_f32_f16<<<(WSZ / 4 + 255) / 256, 256, 0, stream>>>(Wv, w16 + 2 * (size_t)WSZ, WSZ);
  cvt_f32_f16<<<(WSZ / 4 + 255) / 256, 256, 0, stream>>>(Wo, w16 + 3 * (size_t)WSZ, WSZ);

  dim3 gp(64, 16), blk(128);
  gemm_proj<<<gp, blk, 0, stream>>>(hs16, w16 + 0 * (size_t)WSZ, bq, q16, QSCALE, 0);
  gemm_proj<<<gp, blk, 0, stream>>>(hs16, w16 + 1 * (size_t)WSZ, bk, k16, 1.0f, 0);
  gemm_proj<<<gp, blk, 0, stream>>>(hs16, w16 + 2 * (size_t)WSZ, bv, vt16, 1.0f, 1);

  attn_flash<<<dim3(Ssz / 64, Bsz * NHsz), blk, 0, stream>>>(q16, k16, vt16, pb, msk, ao16);

  gemm_out<<<gp, blk, 0, stream>>>(ao16, w16 + 3 * (size_t)WSZ, bo, out);
}
